// InputModule_57612691308660
// MI455X (gfx1250) — matrix operand order repaired and checked by reading, not compiled, not run on hardware
//
#include <hip/hip_runtime.h>

typedef __attribute__((ext_vector_type(16))) _Float16 v16h;
typedef __attribute__((ext_vector_type(8)))  _Float16 v8h;
typedef __attribute__((ext_vector_type(4)))  _Float16 v4h;
typedef __attribute__((ext_vector_type(8)))  float    v8f;

#define FLAT_PITCH 1032   // 1024 + 8 halves padding: row stride 516 dw -> +4 banks/row

// ---------------- kernel 0: Wsq f32 -> f16 (512x1024, K contiguous) ----------
__global__ __launch_bounds__(256) void k_cvt_wsq(const float* __restrict__ src,
                                                 _Float16* __restrict__ dst) {
  int i = (blockIdx.x * 256 + threadIdx.x) * 4;   // grid 512 covers 524288 exactly
  float4 f = *(const float4*)(src + i);
  v4h h;
  h[0] = (_Float16)f.x; h[1] = (_Float16)f.y;
  h[2] = (_Float16)f.z; h[3] = (_Float16)f.w;
  *(v4h*)(dst + i) = h;
}

// ---------------- fused kernel: one block per batch element b ----------------
// Stage 1: rel -> flat (f16, LDS).  Stage 2: WMMA GEMM 64x1024 @ 1024x512.
__global__ __launch_bounds__(256) void k_fused(
    const float*    __restrict__ inpG,    // (B,8,8,8)
    const float*    __restrict__ WrelG,   // (15,15,16,8,8)
    const float*    __restrict__ brelG,   // (15,15,16)
    const _Float16* __restrict__ WsqH,    // (512,1024) f16
    const float*    __restrict__ bsqG,    // (512)
    float*          __restrict__ outG) {  // (B,32,32,32)
  __shared__ float    s_inp[64 * 8];               //   2 KiB
  __shared__ float    s_w[16 * 64];                //   4 KiB  W[da][dc] as [k][i*8+j]
  __shared__ float    s_brel[16];
  __shared__ _Float16 s_flat[64 * FLAT_PITCH];     // 129 KiB

  const int tid = threadIdx.x;
  const int b   = blockIdx.x;

  // load inp[b] : 512 floats, 2 per thread
  *(float2*)&s_inp[tid * 2] = *(const float2*)(inpG + (size_t)b * 512 + tid * 2);

  // ---------------- stage 1 ----------------
  for (int pair = 0; pair < 225; ++pair) {
    __syncthreads();  // protect s_w/s_brel from previous iteration's readers
    *(float4*)&s_w[tid * 4] = *(const float4*)(WrelG + pair * 1024 + tid * 4);
    if (tid < 16) s_brel[tid] = brelG[pair * 16 + tid];
    __syncthreads();

    const int da   = pair / 15;
    const int dc   = pair - da * 15;
    const int na   = 8 - (da < 7 ? 7 - da : da - 7);
    const int nc   = 8 - (dc < 7 ? 7 - dc : dc - 7);
    const int amin = da < 7 ? 7 - da : 0;
    const int cmin = dc < 7 ? 7 - dc : 0;
    const int tup  = na * nc;
    const int total = tup * 16;

    for (int w = tid; w < total; w += 256) {
      const int k  = w / tup;            // k slowest -> W LDS reads broadcast
      const int t  = w - k * tup;
      const int ia = t / nc;
      const int ic = t - ia * nc;
      const int a = amin + ia, c = cmin + ic;
      const int x = a + da - 7, y = c + dc - 7;
      const float* u  = &s_inp[(a * 8 + c) * 8];
      const float* v  = &s_inp[(x * 8 + y) * 8];
      const float* wk = &s_w[k * 64];
      float acc = s_brel[k];
#pragma unroll
      for (int i = 0; i < 8; ++i) {
        float s = 0.f;
#pragma unroll
        for (int j = 0; j < 8; ++j) s = fmaf(wk[i * 8 + j], v[j], s);
        acc = fmaf(u[i], s, acc);
      }
      s_flat[(a * 8 + c) * FLAT_PITCH + x * 128 + y * 16 + k] = (_Float16)acc;
    }
  }
  __syncthreads();

  // ---------------- stage 2: WMMA ----------------
  // wave wv: m-tiles 0..3, n-tiles wv*4 .. wv*4+3  (tiles 16x16, K step 32)
  const int wv    = tid >> 5;
  const int lane  = tid & 31;
  const int lrow  = lane & 15;   // M (A/D) or N (B) within tile
  const int lhalf = lane >> 4;

  v8f acc[4][4] = {};

  for (int kb = 0; kb < 1024; kb += 32) {
    v16h af[4];
#pragma unroll
    for (int mt = 0; mt < 4; ++mt) {
      // A operand in this code's contiguous k order (B matches): lanes 0-15 hold K kb..kb+15,
      // lanes 16-31 hold K kb+16..kb+31; M = lane&15
      const _Float16* p = &s_flat[(mt * 16 + lrow) * FLAT_PITCH + kb + lhalf * 16];
      v8h lo = *(const v8h*)p;
      v8h hi = *(const v8h*)(p + 8);
      af[mt] = __builtin_shufflevector(lo, hi, 0, 1, 2, 3, 4, 5, 6, 7,
                                               8, 9, 10, 11, 12, 13, 14, 15);
    }
    v16h bf[4];
#pragma unroll
    for (int nt = 0; nt < 4; ++nt) {
      // B operand in the same contiguous k order as A: N = lane&15; lanes 0-15: K kb..kb+15,
      // lanes 16-31: K kb+16..kb+31 -> one 32B contiguous load (K-major Wsq)
      const int n = (wv * 4 + nt) * 16 + lrow;
      bf[nt] = *(const v16h*)(WsqH + n * 1024 + kb + lhalf * 16);
    }
#pragma unroll
    for (int mt = 0; mt < 4; ++mt)
#pragma unroll
      for (int nt = 0; nt < 4; ++nt)
        acc[mt][nt] = __builtin_amdgcn_wmma_f32_16x16x32_f16(
            false, af[mt], false, bf[nt], (short)0, acc[mt][nt], false, false);
  }

  // ---------------- epilogue: bias + spatial interleave store ----------------
  // D layout: VGPR r -> M = r + 8*(lane>=16), N = lane&15
#pragma unroll
  for (int nt = 0; nt < 4; ++nt) {
    const int n    = (wv * 4 + nt) * 16 + lrow;       // n = X*128 + Y*32 + o
    const float bias = bsqG[n];
    const int X = n >> 7, Y = (n >> 5) & 3, o = n & 31;
#pragma unroll
    for (int mt = 0; mt < 4; ++mt) {
#pragma unroll
      for (int r = 0; r < 8; ++r) {
        const int m  = mt * 16 + lhalf * 8 + r;       // m = h*8 + w
        const int hh = m >> 3, ww = m & 7;
        outG[(((size_t)b * 32 + (X * 8 + hh)) * 32 + (Y * 8 + ww)) * 32 + o] =
            acc[mt][nt][r] + bias;
      }
    }
  }
}

extern "C" void kernel_launch(void* const* d_in, const int* in_sizes, int n_in,
                              void* d_out, int out_size, void* d_ws, size_t ws_size,
                              hipStream_t stream) {
  (void)in_sizes; (void)n_in; (void)out_size; (void)ws_size;
  const float* inp  = (const float*)d_in[0];   // (1024,8,8,8)
  const float* Wrel = (const float*)d_in[1];   // (15,15,16,8,8)
  const float* brel = (const float*)d_in[2];   // (15,15,16)
  const float* Wsq  = (const float*)d_in[3];   // (4,4,32,1024)
  const float* bsq  = (const float*)d_in[4];   // (4,4,32)
  float* out = (float*)d_out;
  _Float16* WsqH = (_Float16*)d_ws;            // 512*1024*2 = 1 MiB of scratch

  k_cvt_wsq<<<512, 256, 0, stream>>>(Wsq, WsqH);
  k_fused<<<1024, 256, 0, stream>>>(inp, Wrel, brel, WsqH, bsq, out);
}
